// NerfRenderer_62165356642725
// MI455X (gfx1250) — matrix operand order repaired and checked by reading, not compiled, not run on hardware
//
#include <hip/hip_runtime.h>
#include <hip/hip_bf16.h>
#include <math.h>

typedef __attribute__((ext_vector_type(16))) _Float16 v16h;
typedef __attribute__((ext_vector_type(8)))  float    v8f;

#define NRAYS  4096
#define NSAMP  256
#define NTILES 16
#define GRIDN  128
#define FEAT   64
#define HID    128
#define RGBH   64
#define OPTHR  0.01f

__device__ __forceinline__ float warp_t(int i) {
  const float end = 1.0f - 1.0f / (float)(NSAMP + 2);
  float ts = end * (float)i / (float)NSAMP;
  return (ts < 0.5f) ? (2.0f * ts) : (1.0f / (2.0f - 2.0f * ts));
}
__device__ __forceinline__ float sigmoidf_(float x) { return 1.0f / (1.0f + expf(-x)); }
__device__ __forceinline__ float softplusf_(float x) { return (x > 20.0f) ? x : log1pf(expf(x)); }

__device__ __forceinline__ float grid_tri(const float* __restrict__ g, float x, float y, float z) {
  float ix = ((x + 1.0f) * GRIDN - 1.0f) * 0.5f;
  float iy = ((y + 1.0f) * GRIDN - 1.0f) * 0.5f;
  float iz = ((z + 1.0f) * GRIDN - 1.0f) * 0.5f;
  float fx0 = floorf(ix), fy0 = floorf(iy), fz0 = floorf(iz);
  float fx = ix - fx0, fy = iy - fy0, fz = iz - fz0;
  int x0 = (int)fx0, y0 = (int)fy0, z0 = (int)fz0;
  float out = 0.0f;
#pragma unroll
  for (int dz = 0; dz < 2; ++dz)
#pragma unroll
    for (int dy = 0; dy < 2; ++dy)
#pragma unroll
      for (int dx = 0; dx < 2; ++dx) {
        int cx = x0 + dx, cy = y0 + dy, cz = z0 + dz;
        bool valid = (cx >= 0) && (cx < GRIDN) && (cy >= 0) && (cy < GRIDN) && (cz >= 0) && (cz < GRIDN);
        float w = (dx ? fx : 1.0f - fx) * (dy ? fy : 1.0f - fy) * (dz ? fz : 1.0f - fz);
        if (valid) out += w * g[((cz * GRIDN) + cy) * GRIDN + cx];
      }
  return out;
}

extern "C" __global__ __launch_bounds__(32)
void nerf_render_kernel(const float* __restrict__ rays_o, const float* __restrict__ rays_d,
                        const float* __restrict__ grid,
                        const float* __restrict__ f_w1, const float* __restrict__ f_b1,
                        const float* __restrict__ f_w2, const float* __restrict__ f_b2,
                        const float* __restrict__ s_w,  const float* __restrict__ s_b,
                        const float* __restrict__ r_w1, const float* __restrict__ r_b1,
                        const float* __restrict__ r_w2, const float* __restrict__ r_b2,
                        float* __restrict__ out)
{
  // B fragments pre-swizzled into per-lane WMMA layout: [0..15]=f_w2, [16..27]=r_w1 (zero-padded K=96)
  __shared__ alignas(32) _Float16 sh_frag[28][32][16];
  __shared__ float sh_fw1[3 * HID];
  __shared__ float sh_fb1[HID];
  __shared__ float sh_fb2[FEAT];
  __shared__ float sh_sw[FEAT];
  __shared__ float sh_rb1[RGBH];
  __shared__ float sh_rw2[RGBH * 3];
  __shared__ float sh_rb2_sb[4];
  __shared__ alignas(8) _Float16 sh_feat[16][FEAT];
  __shared__ float sh_h[16][RGBH];
  __shared__ float sh_c[16][3];
  __shared__ float sh_occ[16];
  __shared__ float sh_a[16];
  __shared__ float sh_wgt[16];

  const int ray  = blockIdx.x;
  const int lane = threadIdx.x;     // wave32
  const int half = lane >> 4;       // 0: lanes 0-15, 1: lanes 16-31
  const int lr   = lane & 15;

  // ---- stage small weights in LDS ----
  for (int i = lane; i < 3 * HID; i += 32) sh_fw1[i] = f_w1[i];
  for (int i = lane; i < HID; i += 32) sh_fb1[i] = f_b1[i];
  for (int i = lane; i < FEAT; i += 32) { sh_fb2[i] = f_b2[i]; sh_sw[i] = s_w[i]; sh_rb1[i] = r_b1[i]; }
  for (int i = lane; i < RGBH * 3; i += 32) sh_rw2[i] = r_w2[i];
  if (lane < 3) sh_rb2_sb[lane] = r_b2[lane];
  if (lane == 3) sh_rb2_sb[3] = s_b[0];

  // ---- build B fragments (k order used here, same as A: lanes 0-15 hold K 0..15, lanes 16-31 hold K 16..31; 2 halves per VGPR) ----
#pragma unroll
  for (int kt = 0; kt < 4; ++kt)
#pragma unroll
    for (int nt = 0; nt < 4; ++nt) {
      int fi  = kt * 4 + nt;
      int col = nt * 16 + lr;
      int kb  = kt * 32 + half * 16;
#pragma unroll
      for (int v = 0; v < 8; ++v) {
        sh_frag[fi][lane][2 * v]     = (_Float16)f_w2[(kb + 2 * v)     * FEAT + col];
        sh_frag[fi][lane][2 * v + 1] = (_Float16)f_w2[(kb + 2 * v + 1) * FEAT + col];
      }
    }
#pragma unroll
  for (int kt = 0; kt < 3; ++kt)
#pragma unroll
    for (int nt = 0; nt < 4; ++nt) {
      int fi  = 16 + kt * 4 + nt;
      int col = nt * 16 + lr;
      int kb  = kt * 32 + half * 16;
#pragma unroll
      for (int v = 0; v < 8; ++v) {
        int k0 = kb + 2 * v, k1 = kb + 2 * v + 1;
        sh_frag[fi][lane][2 * v]     = (k0 < FEAT + 3) ? (_Float16)r_w1[k0 * RGBH + col] : (_Float16)0.0f;
        sh_frag[fi][lane][2 * v + 1] = (k1 < FEAT + 3) ? (_Float16)r_w1[k1 * RGBH + col] : (_Float16)0.0f;
      }
    }

  const float ox = rays_o[ray * 3 + 0], oy = rays_o[ray * 3 + 1], oz = rays_o[ray * 3 + 2];
  const float dx = rays_d[ray * 3 + 0], dy = rays_d[ray * 3 + 1], dz = rays_d[ray * 3 + 2];

  __syncthreads();

  float logT = 0.0f;
  float acc0 = 0.0f, acc1 = 0.0f, acc2 = 0.0f;

  for (int tile = 0; tile < NTILES; ++tile) {
    // 1) positions -> mip360 contraction -> trilinear occupancy (one sample per lane 0..15)
    if (lane < 16) {
      int s = tile * 16 + lane;
      float t = warp_t(s);
      float px = ox + dx * t, py = oy + dy * t, pz = oz + dz * t;
      float nrm = sqrtf(px * px + py * py + pz * pz);
      float scl = (nrm <= 1.0f) ? 0.5f : (2.0f - 1.0f / nrm) / nrm * 0.5f;
      float cx = px * scl, cy = py * scl, cz = pz * scl;
      sh_c[lane][0] = cx; sh_c[lane][1] = cy; sh_c[lane][2] = cz;
      sh_occ[lane] = grid_tri(grid, cx, cy, cz);
    }
    __syncthreads();

    // 2) layer-1 (K=3, VALU) written straight into 16x32 f16 A-fragment layout
    float c0 = sh_c[lr][0], c1 = sh_c[lr][1], c2 = sh_c[lr][2];
    v16h afr[4];
#pragma unroll
    for (int kt = 0; kt < 4; ++kt) {
      int kb = kt * 32 + half * 16;
#pragma unroll
      for (int v = 0; v < 8; ++v) {
        int k0 = kb + 2 * v;
#pragma unroll
        for (int e = 0; e < 2; ++e) {
          int k = k0 + e;
          float h = fmaf(c0, sh_fw1[0 * HID + k],
                    fmaf(c1, sh_fw1[1 * HID + k],
                    fmaf(c2, sh_fw1[2 * HID + k], sh_fb1[k])));
          afr[kt][2 * v + e] = (_Float16)fmaxf(h, 0.0f);
        }
      }
    }

    // 3) layer-2: feat[16x64] = hidden[16x128] @ f_w2[128x64]  (WMMA f16 -> f32)
#pragma unroll
    for (int nt = 0; nt < 4; ++nt) {
      v8f acc = {};
#pragma unroll
      for (int kt = 0; kt < 4; ++kt) {
        v16h b = *(const v16h*)&sh_frag[kt * 4 + nt][lane][0];
        acc = __builtin_amdgcn_wmma_f32_16x16x32_f16(false, afr[kt], false, b, (short)0, acc, false, false);
      }
      int col = nt * 16 + lr;
      float bias = sh_fb2[col];
#pragma unroll
      for (int v = 0; v < 8; ++v) {
        int row = v + half * 8;
        float val = acc[v] + bias;
        val = (sh_occ[row] > OPTHR) ? val : 0.0f;   // masked feature write
        sh_feat[row][col] = (_Float16)val;
      }
    }
    __syncthreads();

    // 4) sigma = softplus(feat @ s_w + s_b), masked; a = -sigma*dist
    if (lane < 16) {
      float s = sh_rb2_sb[3];
#pragma unroll 8
      for (int c = 0; c < FEAT; ++c) s += (float)sh_feat[lane][c] * sh_sw[c];
      float sigma = (sh_occ[lane] > OPTHR) ? softplusf_(s) : 0.0f;
      int smp = tile * 16 + lane;
      float d = warp_t(smp + 1) - warp_t(smp);
      sh_a[lane] = -sigma * d;
    }
    __syncthreads();

    // 5) transmittance prefix within tile; uniform logT update
    float total = 0.0f;
    {
      float pref = 0.0f;
#pragma unroll
      for (int i = 0; i < 16; ++i) {
        float ai = sh_a[i];
        if (i < lr) pref += ai;
        total += ai;
      }
      if (lane < 16) {
        float tr = expf(logT + pref);
        float w  = tr * (1.0f - expf(sh_a[lr]));
        bool m2  = (sh_occ[lr] > OPTHR) && (tr > 1e-4f);
        sh_wgt[lr] = m2 ? w : 0.0f;
      }
    }
    logT += total;
    __syncthreads();

    // 6) rgb hidden: h = relu([feat | dir | 0](16x96) @ r_w1(96x64))  (WMMA)
    v16h rafr[3];
#pragma unroll
    for (int kt = 0; kt < 3; ++kt) {
      int kb = kt * 32 + half * 16;
#pragma unroll
      for (int v = 0; v < 8; ++v) {
        int k0 = kb + 2 * v;
#pragma unroll
        for (int e = 0; e < 2; ++e) {
          int k = k0 + e;
          _Float16 val;
          if (k < FEAT)            val = sh_feat[lr][k];
          else if (k == FEAT)      val = (_Float16)dx;
          else if (k == FEAT + 1)  val = (_Float16)dy;
          else if (k == FEAT + 2)  val = (_Float16)dz;
          else                     val = (_Float16)0.0f;
          rafr[kt][2 * v + e] = val;
        }
      }
    }
#pragma unroll
    for (int nt = 0; nt < 4; ++nt) {
      v8f acc = {};
#pragma unroll
      for (int kt = 0; kt < 3; ++kt) {
        v16h b = *(const v16h*)&sh_frag[16 + kt * 4 + nt][lane][0];
        acc = __builtin_amdgcn_wmma_f32_16x16x32_f16(false, rafr[kt], false, b, (short)0, acc, false, false);
      }
      int col = nt * 16 + lr;
      float bias = sh_rb1[col];
#pragma unroll
      for (int v = 0; v < 8; ++v) {
        int row = v + half * 8;
        sh_h[row][col] = fmaxf(acc[v] + bias, 0.0f);
      }
    }
    __syncthreads();

    // 7) rgb = sigmoid(h @ r_w2 + r_b2) * weight, accumulated per lane
    if (lane < 16) {
      float w = sh_wgt[lane];
      if (w != 0.0f) {
        float o0 = sh_rb2_sb[0], o1 = sh_rb2_sb[1], o2 = sh_rb2_sb[2];
#pragma unroll 8
        for (int c = 0; c < RGBH; ++c) {
          float hv = sh_h[lane][c];
          o0 = fmaf(hv, sh_rw2[c * 3 + 0], o0);
          o1 = fmaf(hv, sh_rw2[c * 3 + 1], o1);
          o2 = fmaf(hv, sh_rw2[c * 3 + 2], o2);
        }
        acc0 += w * sigmoidf_(o0);
        acc1 += w * sigmoidf_(o1);
        acc2 += w * sigmoidf_(o2);
      }
    }
    __syncthreads();

    // early ray termination: trans is non-increasing; all later samples are masked out
    if (logT <= -9.2103403719761827f) break;
  }

  // reduce 16 partial colors across the wave (lanes >=16 hold zero)
#pragma unroll
  for (int off = 16; off >= 1; off >>= 1) {
    acc0 += __shfl_xor(acc0, off, 32);
    acc1 += __shfl_xor(acc1, off, 32);
    acc2 += __shfl_xor(acc2, off, 32);
  }
  if (lane == 0) {
    out[ray * 3 + 0] = acc0;
    out[ray * 3 + 1] = acc1;
    out[ray * 3 + 2] = acc2;
  }
}

extern "C" void kernel_launch(void* const* d_in, const int* in_sizes, int n_in,
                              void* d_out, int out_size, void* d_ws, size_t ws_size,
                              hipStream_t stream) {
  (void)in_sizes; (void)n_in; (void)out_size; (void)d_ws; (void)ws_size;
  const float* rays_o = (const float*)d_in[0];
  const float* rays_d = (const float*)d_in[1];
  const float* grid   = (const float*)d_in[2];
  const float* f_w1   = (const float*)d_in[3];
  const float* f_b1   = (const float*)d_in[4];
  const float* f_w2   = (const float*)d_in[5];
  const float* f_b2   = (const float*)d_in[6];
  const float* s_w    = (const float*)d_in[7];
  const float* s_b    = (const float*)d_in[8];
  const float* r_w1   = (const float*)d_in[9];
  const float* r_b1   = (const float*)d_in[10];
  const float* r_w2   = (const float*)d_in[11];
  const float* r_b2   = (const float*)d_in[12];
  float* out = (float*)d_out;

  nerf_render_kernel<<<NRAYS, 32, 0, stream>>>(
      rays_o, rays_d, grid, f_w1, f_b1, f_w2, f_b2, s_w, s_b,
      r_w1, r_b1, r_w2, r_b2, out);
}
